// Attn_57415122812960
// MI455X (gfx1250) — matrix operand order repaired and checked by reading, not compiled, not run on hardware
//
#include <hip/hip_runtime.h>
#include <math.h>

// Problem sizes (fixed by the reference)
#define NBATCH 2
#define CH     256     // full channel count
#define C2     128     // C/2
#define NEDGE  65536   // N (edges)
#define KMESH  640     // 128 channels * 5 symmetric taps
#define NT     32      // N-tile per block (2 x 16-wide WMMA subtiles)
#define EPSV   1e-5f

typedef __attribute__((ext_vector_type(16))) __bf16       v16bf;
typedef __attribute__((ext_vector_type(8)))  float        v8f;
typedef __attribute__((ext_vector_type(4)))  unsigned int u32x4;

struct Frag32B { u32x4 lo, hi; };   // 32 bytes == 16 bf16

static __device__ __forceinline__ unsigned short f2bf(float f) {
  unsigned int u = __builtin_bit_cast(unsigned int, f);
  u += 0x7FFFu + ((u >> 16) & 1u);            // round-to-nearest-even
  return (unsigned short)(u >> 16);
}
static __device__ __forceinline__ float bf2f(unsigned short h) {
  return __builtin_bit_cast(float, (unsigned int)h << 16);
}

static __device__ __forceinline__ v8f wmma_bf16(v16bf a, v16bf b, v8f c) {
  // D = A(16x32) * B(32x16) + C(16x16), fp32 accumulate
  return __builtin_amdgcn_wmma_f32_16x16x32_bf16(false, a, false, b, (short)0, c,
                                                 false, false);
}

// A fragment: weights stored bf16 row-major (Cout x K).
// Lane L: row M = m0 + L%16; K chunks [k0 + (L/16)*8 .. +7] and [+16 .. +23].
static __device__ __forceinline__ v16bf load_a_frag(const unsigned short* W, int ldk,
                                                    int m0, int k0, int lane) {
  const unsigned short* p = W + (size_t)(m0 + (lane & 15)) * ldk
                              + (size_t)(k0 + ((lane >> 4) << 3));
  Frag32B f;
  f.lo = *(const u32x4*)(p);        // K = base .. base+7
  f.hi = *(const u32x4*)(p + 16);   // K = base+16 .. base+23
  return __builtin_bit_cast(v16bf, f);
}

// B fragment: activations stored bf16 N-major (row n has ldk contiguous channels).
// Lane L: col N = n0 + L%16; K chunks [k0 + (L/16)*8 .. +7] and [+16 .. +23].
static __device__ __forceinline__ v16bf load_b_frag(const unsigned short* X, int ldk,
                                                    int n0, int k0, int lane) {
  const unsigned short* p = X + (size_t)(n0 + (lane & 15)) * ldk
                              + (size_t)(k0 + ((lane >> 4) << 3));
  Frag32B f;
  f.lo = *(const u32x4*)(p);
  f.hi = *(const u32x4*)(p + 16);
  return __builtin_bit_cast(v16bf, f);
}

static __device__ __forceinline__ v8f add_bias(v8f c, const float* bias, int m0, int lane) {
  int cb = m0 + ((lane >> 4) << 3);
#pragma unroll
  for (int v = 0; v < 8; ++v) c[v] += bias[cb + v];
  return c;
}

// C/D tile 16x16 f32 -> packed bf16 store into (N, C) layout: one 16B store/lane.
static __device__ __forceinline__ void store_tile_bf16(unsigned short* dst, int ldc,
                                                       int n0, int c0, int lane, v8f c) {
  int n  = n0 + (lane & 15);
  int cb = c0 + ((lane >> 4) << 3);
  u32x4 p;
  p.x = (unsigned)f2bf(c[0]) | ((unsigned)f2bf(c[1]) << 16);
  p.y = (unsigned)f2bf(c[2]) | ((unsigned)f2bf(c[3]) << 16);
  p.z = (unsigned)f2bf(c[4]) | ((unsigned)f2bf(c[5]) << 16);
  p.w = (unsigned)f2bf(c[6]) | ((unsigned)f2bf(c[7]) << 16);
  *(u32x4*)(dst + (size_t)n * ldc + cb) = p;
}

// Per-channel sum / sum-of-squares over the 16 n-columns of a tile (for instance norm).
static __device__ __forceinline__ void tile_stats(v8f c, float* sum, float* sumsq,
                                                  int b, int m0, int lane) {
#pragma unroll
  for (int v = 0; v < 8; ++v) {
    float s = c[v];
    float q = c[v] * c[v];
#pragma unroll
    for (int m = 1; m < 16; m <<= 1) {      // reduce across the 16 n-lanes
      s += __shfl_xor(s, m, 32);
      q += __shfl_xor(q, m, 32);
    }
    if ((lane & 15) == 0) {
      int ch = m0 + ((lane >> 4) << 3) + v;
      atomicAdd(&sum[b * CH + ch], s);
      atomicAdd(&sumsq[b * CH + ch], q);
    }
  }
}

// ---------------------------------------------------------------------------
// Kernel: fp32 -> bf16 weight conversion
__global__ __launch_bounds__(256) void k_cvt(const float* __restrict__ src,
                                             unsigned short* __restrict__ dst, int n) {
  int i = blockIdx.x * 256 + threadIdx.x;
  if (i < n) dst[i] = f2bf(src[i]);
}

__global__ __launch_bounds__(256) void k_zero(float* __restrict__ p, int n) {
  int i = blockIdx.x * 256 + threadIdx.x;
  if (i < n) p[i] = 0.0f;
}

// ---------------------------------------------------------------------------
// Kernel 1: conv_a(x0) || conv_b(x1)  ->  x_all (B, N, 128) bf16
// Stages fp32 (C,N) tiles through LDS with transpose + bf16 convert.
__global__ __launch_bounds__(256) void k_convAB(const float* __restrict__ x0,
                                                const float* __restrict__ x1,
                                                const unsigned short* __restrict__ WaB,
                                                const unsigned short* __restrict__ WbB,
                                                const float* __restrict__ ba,
                                                const float* __restrict__ bb,
                                                unsigned short* __restrict__ xallT) {
  __shared__ unsigned short lds0[NT][32];   // x0 tile, [n][k]
  __shared__ unsigned short lds1[NT][32];   // x1 tile, [n][k]
  const int b    = blockIdx.y;
  const int n0   = blockIdx.x * NT;
  const int t    = threadIdx.x;
  const int lane = t & 31;
  const int wave = t >> 5;
  const bool isA = wave < 4;
  const int  m0  = (isA ? wave : wave - 4) * 16;

  const float* X0 = x0 + (size_t)b * CH * NEDGE;
  const float* X1 = x1 + (size_t)b * CH * NEDGE;
  const unsigned short* W = isA ? WaB : WbB;

  v8f acc0 = {}, acc1 = {};
  for (int k0 = 0; k0 < CH; k0 += 32) {
    __syncthreads();
    for (int idx = t; idx < 32 * NT; idx += 256) {   // 32 k-rows x NT n-cols
      int k = idx >> 5, n = idx & (NT - 1);
      lds0[n][k] = f2bf(X0[(size_t)(k0 + k) * NEDGE + n0 + n]);
      lds1[n][k] = f2bf(X1[(size_t)(k0 + k) * NEDGE + n0 + n]);
    }
    __syncthreads();
    const unsigned short* bp = isA ? &lds0[0][0] : &lds1[0][0];
    v16bf af  = load_a_frag(W, CH, m0, k0, lane);     // reused by both subtiles
    v16bf bf0 = load_b_frag(bp, 32, 0, 0, lane);
    v16bf bf1 = load_b_frag(bp, 32, 16, 0, lane);
    acc0 = wmma_bf16(af, bf0, acc0);
    acc1 = wmma_bf16(af, bf1, acc1);
  }
  const float* bias = isA ? ba : bb;
  acc0 = add_bias(acc0, bias, m0, lane);
  acc1 = add_bias(acc1, bias, m0, lane);
  unsigned short* dst = xallT + (size_t)b * NEDGE * C2;
  const int c0 = (isA ? 0 : 64) + m0;
  store_tile_bf16(dst, C2, n0,      c0, lane, acc0);
  store_tile_bf16(dst, C2, n0 + 16, c0, lane, acc1);
}

// ---------------------------------------------------------------------------
// Kernel 2: a_local / b_local (128->128)  -> low halves of x_a_two / x_b_two
__global__ __launch_bounds__(256) void k_local(const unsigned short* __restrict__ xallT,
                                               const unsigned short* __restrict__ WalB,
                                               const unsigned short* __restrict__ WblB,
                                               const float* __restrict__ bal,
                                               const float* __restrict__ bbl,
                                               unsigned short* __restrict__ xa2T,
                                               unsigned short* __restrict__ xb2T) {
  const int b    = blockIdx.y;
  const int n0   = blockIdx.x * NT;
  const int t    = threadIdx.x;
  const int lane = t & 31;
  const int m0   = (t >> 5) * 16;          // 8 waves cover Cout=128
  const unsigned short* X = xallT + (size_t)b * NEDGE * C2;

  v8f cA0 = {}, cA1 = {}, cB0 = {}, cB1 = {};
  for (int k0 = 0; k0 < C2; k0 += 32) {
    v16bf aA  = load_a_frag(WalB, C2, m0, k0, lane);  // A frags reused by 2 subtiles
    v16bf aB  = load_a_frag(WblB, C2, m0, k0, lane);
    v16bf bf0 = load_b_frag(X, C2, n0,      k0, lane); // B frags shared by 2 convs
    v16bf bf1 = load_b_frag(X, C2, n0 + 16, k0, lane);
    cA0 = wmma_bf16(aA, bf0, cA0);
    cA1 = wmma_bf16(aA, bf1, cA1);
    cB0 = wmma_bf16(aB, bf0, cB0);
    cB1 = wmma_bf16(aB, bf1, cB1);
  }
  cA0 = add_bias(cA0, bal, m0, lane);
  cA1 = add_bias(cA1, bal, m0, lane);
  cB0 = add_bias(cB0, bbl, m0, lane);
  cB1 = add_bias(cB1, bbl, m0, lane);
  unsigned short* Da = xa2T + (size_t)b * NEDGE * CH;
  unsigned short* Db = xb2T + (size_t)b * NEDGE * CH;
  store_tile_bf16(Da, CH, n0,      m0, lane, cA0);
  store_tile_bf16(Da, CH, n0 + 16, m0, lane, cA1);
  store_tile_bf16(Db, CH, n0,      m0, lane, cB0);
  store_tile_bf16(Db, CH, n0 + 16, m0, lane, cB1);
}

// ---------------------------------------------------------------------------
// Kernel 3: MeshConv for both a_tri and b_tri. Build symmetric feature tile g
// (K=640 x NT edges) once in LDS, GEMM against Wat and Wbt.
__global__ __launch_bounds__(256) void k_mesh(const unsigned short* __restrict__ xallT,
                                              const int* __restrict__ gemm,
                                              const unsigned short* __restrict__ WatB,
                                              const unsigned short* __restrict__ WbtB,
                                              const float* __restrict__ bat,
                                              const float* __restrict__ bbt,
                                              unsigned short* __restrict__ xa2T,
                                              unsigned short* __restrict__ xb2T) {
  __shared__ unsigned short g[NT][KMESH];   // [edge][c*5 + tap], 40 KB
  const int b    = blockIdx.y;
  const int n0   = blockIdx.x * NT;
  const int t    = threadIdx.x;
  const int lane = t & 31;
  const int m0   = (t >> 5) * 16;           // 8 waves cover Cout=128
  const unsigned short* X = xallT + (size_t)b * NEDGE * C2;
  const int* G = gemm + ((size_t)b * NEDGE + n0) * 4;

  // Gather + symmetric functions (gathers stay in 192MB L2: x_all is 33MB bf16)
  for (int idx = t; idx < NT * C2; idx += 256) {
    int e = idx >> 7, c = idx & 127;
    int i0 = G[e * 4 + 0], i1 = G[e * 4 + 1], i2 = G[e * 4 + 2], i3 = G[e * 4 + 3];
    float f0 = bf2f(X[(size_t)(n0 + e) * C2 + c]);
    float f1 = bf2f(X[(size_t)i0 * C2 + c]);
    float f2 = bf2f(X[(size_t)i1 * C2 + c]);
    float f3 = bf2f(X[(size_t)i2 * C2 + c]);
    float f4 = bf2f(X[(size_t)i3 * C2 + c]);
    unsigned short* gp = &g[e][c * 5];
    gp[0] = f2bf(f0);
    gp[1] = f2bf(f1 + f3);
    gp[2] = f2bf(f2 + f4);
    gp[3] = f2bf(fabsf(f1 - f3));
    gp[4] = f2bf(fabsf(f2 - f4));
  }
  __syncthreads();

  const unsigned short* gb = &g[0][0];
  v8f cA0 = {}, cA1 = {}, cB0 = {}, cB1 = {};
  for (int k0 = 0; k0 < KMESH; k0 += 32) {
    v16bf aA  = load_a_frag(WatB, KMESH, m0, k0, lane);
    v16bf aB  = load_a_frag(WbtB, KMESH, m0, k0, lane);
    v16bf bf0 = load_b_frag(gb, KMESH, 0,  k0, lane);
    v16bf bf1 = load_b_frag(gb, KMESH, 16, k0, lane);
    cA0 = wmma_bf16(aA, bf0, cA0);
    cA1 = wmma_bf16(aA, bf1, cA1);
    cB0 = wmma_bf16(aB, bf0, cB0);
    cB1 = wmma_bf16(aB, bf1, cB1);
  }
  cA0 = add_bias(cA0, bat, m0, lane);
  cA1 = add_bias(cA1, bat, m0, lane);
  cB0 = add_bias(cB0, bbt, m0, lane);
  cB1 = add_bias(cB1, bbt, m0, lane);
  unsigned short* Da = xa2T + (size_t)b * NEDGE * CH;
  unsigned short* Db = xb2T + (size_t)b * NEDGE * CH;
  store_tile_bf16(Da, CH, n0,      C2 + m0, lane, cA0);
  store_tile_bf16(Da, CH, n0 + 16, C2 + m0, lane, cA1);
  store_tile_bf16(Db, CH, n0,      C2 + m0, lane, cB0);
  store_tile_bf16(Db, CH, n0 + 16, C2 + m0, lane, cB1);
}

// ---------------------------------------------------------------------------
// Kernel 4: fuse convs (256->256) for both paths + instance-norm statistics.
__global__ __launch_bounds__(256) void k_fuse(const unsigned short* __restrict__ xa2T,
                                              const unsigned short* __restrict__ xb2T,
                                              const unsigned short* __restrict__ WafB,
                                              const unsigned short* __restrict__ WbfB,
                                              const float* __restrict__ baf,
                                              const float* __restrict__ bbf,
                                              unsigned short* __restrict__ yaT,
                                              unsigned short* __restrict__ ybT,
                                              float* __restrict__ sa, float* __restrict__ qa,
                                              float* __restrict__ sb, float* __restrict__ qb) {
  const int b    = blockIdx.y;
  const int n0   = blockIdx.x * NT;
  const int t    = threadIdx.x;
  const int lane = t & 31;
  const int m0   = (t >> 5) * 16;    // wave handles M-tiles m0 and m0+128
  const int m1   = m0 + 128;
  const unsigned short* Xa = xa2T + (size_t)b * NEDGE * CH;
  const unsigned short* Xb = xb2T + (size_t)b * NEDGE * CH;

  v8f ca00 = {}, ca01 = {}, ca10 = {}, ca11 = {};
  v8f cb00 = {}, cb01 = {}, cb10 = {}, cb11 = {};
  for (int k0 = 0; k0 < CH; k0 += 32) {
    if (k0 + 32 < CH) {    // hint the next K-panel into cache (global_prefetch_b8)
      __builtin_prefetch(Xa + (size_t)(n0 + (lane & 15)) * CH + k0 + 32, 0, 1);
      __builtin_prefetch(Xb + (size_t)(n0 + (lane & 15)) * CH + k0 + 32, 0, 1);
    }
    v16bf w0   = load_a_frag(WafB, CH, m0, k0, lane);   // 4 A frags feed 8 WMMAs
    v16bf w1   = load_a_frag(WafB, CH, m1, k0, lane);
    v16bf w2   = load_a_frag(WbfB, CH, m0, k0, lane);
    v16bf w3   = load_a_frag(WbfB, CH, m1, k0, lane);
    v16bf bfa0 = load_b_frag(Xa, CH, n0,      k0, lane);
    v16bf bfa1 = load_b_frag(Xa, CH, n0 + 16, k0, lane);
    v16bf bfb0 = load_b_frag(Xb, CH, n0,      k0, lane);
    v16bf bfb1 = load_b_frag(Xb, CH, n0 + 16, k0, lane);
    ca00 = wmma_bf16(w0, bfa0, ca00);
    ca01 = wmma_bf16(w0, bfa1, ca01);
    ca10 = wmma_bf16(w1, bfa0, ca10);
    ca11 = wmma_bf16(w1, bfa1, ca11);
    cb00 = wmma_bf16(w2, bfb0, cb00);
    cb01 = wmma_bf16(w2, bfb1, cb01);
    cb10 = wmma_bf16(w3, bfb0, cb10);
    cb11 = wmma_bf16(w3, bfb1, cb11);
  }
  ca00 = add_bias(ca00, baf, m0, lane);
  ca01 = add_bias(ca01, baf, m0, lane);
  ca10 = add_bias(ca10, baf, m1, lane);
  ca11 = add_bias(ca11, baf, m1, lane);
  cb00 = add_bias(cb00, bbf, m0, lane);
  cb01 = add_bias(cb01, bbf, m0, lane);
  cb10 = add_bias(cb10, bbf, m1, lane);
  cb11 = add_bias(cb11, bbf, m1, lane);

  unsigned short* Ya = yaT + (size_t)b * NEDGE * CH;
  unsigned short* Yb = ybT + (size_t)b * NEDGE * CH;
  store_tile_bf16(Ya, CH, n0,      m0, lane, ca00);
  store_tile_bf16(Ya, CH, n0 + 16, m0, lane, ca01);
  store_tile_bf16(Ya, CH, n0,      m1, lane, ca10);
  store_tile_bf16(Ya, CH, n0 + 16, m1, lane, ca11);
  store_tile_bf16(Yb, CH, n0,      m0, lane, cb00);
  store_tile_bf16(Yb, CH, n0 + 16, m0, lane, cb01);
  store_tile_bf16(Yb, CH, n0,      m1, lane, cb10);
  store_tile_bf16(Yb, CH, n0 + 16, m1, lane, cb11);

  tile_stats(ca00, sa, qa, b, m0, lane);
  tile_stats(ca01, sa, qa, b, m0, lane);
  tile_stats(ca10, sa, qa, b, m1, lane);
  tile_stats(ca11, sa, qa, b, m1, lane);
  tile_stats(cb00, sb, qb, b, m0, lane);
  tile_stats(cb01, sb, qb, b, m0, lane);
  tile_stats(cb10, sb, qb, b, m1, lane);
  tile_stats(cb11, sb, qb, b, m1, lane);
}

// ---------------------------------------------------------------------------
// Kernel 5: instance norm + sigmoid + 2-way softmax (= sigmoid(wa-wb)) + blend.
__global__ __launch_bounds__(256) void k_final(const float* __restrict__ x0,
                                               const float* __restrict__ x1,
                                               const unsigned short* __restrict__ yaT,
                                               const unsigned short* __restrict__ ybT,
                                               const float* __restrict__ sa,
                                               const float* __restrict__ qa,
                                               const float* __restrict__ sb,
                                               const float* __restrict__ qb,
                                               float* __restrict__ out) {
  const int b = blockIdx.y;
  const int n = blockIdx.x;
  const int c = threadIdx.x;              // 256 channels
  const float invN = 1.0f / (float)NEDGE;

  float ma = sa[b * CH + c] * invN;
  float va = qa[b * CH + c] * invN - ma * ma;
  float mb = sb[b * CH + c] * invN;
  float vb = qb[b * CH + c] * invN - mb * mb;

  size_t ti = ((size_t)b * NEDGE + n) * CH + c;
  float ya = bf2f(yaT[ti]);
  float yb = bf2f(ybT[ti]);
  float wa = 1.0f / (1.0f + __expf(-(ya - ma) * rsqrtf(va + EPSV)));
  float wb = 1.0f / (1.0f + __expf(-(yb - mb) * rsqrtf(vb + EPSV)));
  float w0 = 1.0f / (1.0f + __expf(wb - wa));   // softmax over {wa, wb}

  size_t o = ((size_t)b * CH + c) * NEDGE + n;
  out[o] = x0[o] * w0 + x1[o] * (1.0f - w0);
}

// ---------------------------------------------------------------------------
extern "C" void kernel_launch(void* const* d_in, const int* in_sizes, int n_in,
                              void* d_out, int out_size, void* d_ws, size_t ws_size,
                              hipStream_t stream) {
  (void)in_sizes; (void)n_in; (void)out_size; (void)ws_size;
  const float* x0  = (const float*)d_in[0];
  const float* x1  = (const float*)d_in[1];
  const int*   gm  = (const int*)d_in[2];
  const float* Wa  = (const float*)d_in[3];  const float* ba  = (const float*)d_in[4];
  const float* Wb  = (const float*)d_in[5];  const float* bb  = (const float*)d_in[6];
  const float* Wal = (const float*)d_in[7];  const float* bal = (const float*)d_in[8];
  const float* Wbl = (const float*)d_in[9];  const float* bbl = (const float*)d_in[10];
  const float* Wat = (const float*)d_in[11]; const float* bat = (const float*)d_in[12];
  const float* Wbt = (const float*)d_in[13]; const float* bbt = (const float*)d_in[14];
  const float* Waf = (const float*)d_in[15]; const float* baf = (const float*)d_in[16];
  const float* Wbf = (const float*)d_in[17]; const float* bbf = (const float*)d_in[18];
  float* out = (float*)d_out;

  char* ws = (char*)d_ws;
  size_t off = 0;
  auto carve = [&](size_t bytes) -> char* {
    char* p = ws + off;
    off += (bytes + 255) & ~(size_t)255;
    return p;
  };

  unsigned short* WaB   = (unsigned short*)carve((size_t)64  * 256 * 2);
  unsigned short* WbB   = (unsigned short*)carve((size_t)64  * 256 * 2);
  unsigned short* WalB  = (unsigned short*)carve((size_t)128 * 128 * 2);
  unsigned short* WblB  = (unsigned short*)carve((size_t)128 * 128 * 2);
  unsigned short* WatB  = (unsigned short*)carve((size_t)128 * KMESH * 2);
  unsigned short* WbtB  = (unsigned short*)carve((size_t)128 * KMESH * 2);
  unsigned short* WafB  = (unsigned short*)carve((size_t)256 * 256 * 2);
  unsigned short* WbfB  = (unsigned short*)carve((size_t)256 * 256 * 2);
  unsigned short* xallT = (unsigned short*)carve((size_t)NBATCH * NEDGE * C2 * 2);
  unsigned short* xa2T  = (unsigned short*)carve((size_t)NBATCH * NEDGE * CH * 2);
  unsigned short* xb2T  = (unsigned short*)carve((size_t)NBATCH * NEDGE * CH * 2);
  unsigned short* yaT   = (unsigned short*)carve((size_t)NBATCH * NEDGE * CH * 2);
  unsigned short* ybT   = (unsigned short*)carve((size_t)NBATCH * NEDGE * CH * 2);
  float* sa = (float*)carve((size_t)NBATCH * CH * 4);
  float* qa = (float*)carve((size_t)NBATCH * CH * 4);
  float* sb = (float*)carve((size_t)NBATCH * CH * 4);
  float* qb = (float*)carve((size_t)NBATCH * CH * 4);

  // Weight conversion to bf16
  k_cvt<<<(64 * 256 + 255) / 256, 256, 0, stream>>>(Wa, WaB, 64 * 256);
  k_cvt<<<(64 * 256 + 255) / 256, 256, 0, stream>>>(Wb, WbB, 64 * 256);
  k_cvt<<<(128 * 128 + 255) / 256, 256, 0, stream>>>(Wal, WalB, 128 * 128);
  k_cvt<<<(128 * 128 + 255) / 256, 256, 0, stream>>>(Wbl, WblB, 128 * 128);
  k_cvt<<<(128 * KMESH + 255) / 256, 256, 0, stream>>>(Wat, WatB, 128 * KMESH);
  k_cvt<<<(128 * KMESH + 255) / 256, 256, 0, stream>>>(Wbt, WbtB, 128 * KMESH);
  k_cvt<<<(256 * 256 + 255) / 256, 256, 0, stream>>>(Waf, WafB, 256 * 256);
  k_cvt<<<(256 * 256 + 255) / 256, 256, 0, stream>>>(Wbf, WbfB, 256 * 256);

  // Zero instance-norm accumulators (ws is poisoned by the harness)
  k_zero<<<(NBATCH * CH + 255) / 256, 256, 0, stream>>>(sa, NBATCH * CH);
  k_zero<<<(NBATCH * CH + 255) / 256, 256, 0, stream>>>(qa, NBATCH * CH);
  k_zero<<<(NBATCH * CH + 255) / 256, 256, 0, stream>>>(sb, NBATCH * CH);
  k_zero<<<(NBATCH * CH + 255) / 256, 256, 0, stream>>>(qb, NBATCH * CH);

  dim3 grid(NEDGE / NT, NBATCH);
  k_convAB<<<grid, 256, 0, stream>>>(x0, x1, WaB, WbB, ba, bb, xallT);
  k_local <<<grid, 256, 0, stream>>>(xallT, WalB, WblB, bal, bbl, xa2T, xb2T);
  k_mesh  <<<grid, 256, 0, stream>>>(xallT, gm, WatB, WbtB, bat, bbt, xa2T, xb2T);
  k_fuse  <<<grid, 256, 0, stream>>>(xa2T, xb2T, WafB, WbfB, baf, bbf,
                                     yaT, ybT, sa, qa, sb, qb);
  dim3 gridF(NEDGE, NBATCH);
  k_final <<<gridF, 256, 0, stream>>>(x0, x1, yaT, ybT, sa, qa, sb, qb, out);
}
